// GAE_P_53214644798191
// MI455X (gfx1250) — matrix operand order repaired and checked by reading, not compiled, not run on hardware
//
#include <hip/hip_runtime.h>
#include <hip/hip_bf16.h>

typedef __bf16 bf16_t;
typedef __attribute__((ext_vector_type(16))) __bf16 v16bf;
typedef __attribute__((ext_vector_type(8)))  __bf16 v8bf;
typedef __attribute__((ext_vector_type(8)))  float  v8f;

#define WPB 8  // waves per block (blockDim = 32 x 8 = 256 threads = 8 wave32)
#define JT  4  // j-tiles per wave in the decode kernel

// ---------------------------------------------------------------------------
// Generic 16x16-tile bf16 WMMA GEMM: C[MxN] = op(A[MxK]) @ B[KxN], fp32 in/out,
// bf16 inside the tensor op. One wave per 16x16 C tile, K stepped by 32.
// Optional fused ReLU applied to A while converting fp32 -> bf16.
// ---------------------------------------------------------------------------
template <bool RELUA>
__global__ void gemm16_bf16_wmma(const float* __restrict__ A,
                                 const float* __restrict__ B,
                                 float* __restrict__ C,
                                 int M, int N, int K) {
  const unsigned lane   = threadIdx.x;
  const unsigned waveId = blockIdx.x * WPB + threadIdx.y;
  const unsigned ntiles = (unsigned)N >> 4;
  const unsigned itile  = waveId / ntiles;
  const unsigned jtile  = waveId - itile * ntiles;
  if (itile >= ((unsigned)M >> 4)) return;  // wave-uniform (EXEC stays all-ones)

  const unsigned half = lane >> 4;
  const unsigned l15  = lane & 15;
  const unsigned arow = itile * 16 + l15;   // A: both lane halves map M=0..15
  const unsigned bcol = jtile * 16 + l15;   // B/C/D: row striped across lanes

  v8f acc = {};
  for (int kk = 0; kk < K; kk += 32) {
    // A fragment (16x32 bf16): elem t   <-> K = kk + 8*half + t
    //                          elem 8+t <-> K = kk + 16 + 8*half + t
    v16bf a;
    const float* arp = A + arow * (unsigned)K + kk;
#pragma unroll
    for (int t = 0; t < 8; ++t) {
      float x0 = arp[half * 8 + t];
      float x1 = arp[16 + half * 8 + t];
      if (RELUA) { x0 = fmaxf(x0, 0.0f); x1 = fmaxf(x1, 0.0f); }
      a[t]     = (bf16_t)x0;
      a[8 + t] = (bf16_t)x1;
    }
    // B fragment (32x16 bf16): elem t <-> K = kk + 8*half + t, elem 8+t <-> K = kk + 16 + 8*half + t, col = bcol
    v16bf b;
    const float* bp = B + (unsigned)(kk + half * 8) * (unsigned)N + bcol;
#pragma unroll
    for (int t = 0; t < 16; ++t) b[t] = (bf16_t)bp[(unsigned)(t < 8 ? t : t + 8) * (unsigned)N];

    acc = __builtin_amdgcn_wmma_f32_16x16x32_bf16(false, a, false, b,
                                                  (short)0, acc, false, false);
  }
  // C/D layout: elem r <-> (M = itile*16 + 8*half + r, N = bcol)
  float* crow = C + (itile * 16 + half * 8) * (unsigned)N + bcol;
#pragma unroll
  for (int r = 0; r < 8; ++r) crow[(unsigned)r * (unsigned)N] = acc[r];
}

// ---------------------------------------------------------------------------
// agg[dst[e]] += w[e] * Hin[src[e]]  — one thread per (edge, feature).
// Consecutive threads cover consecutive features of one edge -> coalesced
// gather; scatter uses native fp32 atomics (global_atomic_add_f32).
// ---------------------------------------------------------------------------
template <int LOGF>
__global__ void spmm_scatter(const long long* __restrict__ src,
                             const long long* __restrict__ dst,
                             const float* __restrict__ w,
                             const float* __restrict__ Hin,
                             float* __restrict__ Hout, int E) {
  const int F = 1 << LOGF;
  const long long g = (long long)blockIdx.x * blockDim.x + threadIdx.x;
  const int e = (int)(g >> LOGF);
  const int f = (int)(g & (F - 1));
  if (e >= E) return;
  const int s = (int)src[e];
  const int d = (int)dst[e];
  const float v = w[e] * Hin[(unsigned)s * (unsigned)F + f];
  unsafeAtomicAdd(&Hout[(unsigned)d * (unsigned)F + f], v);
}

__global__ void zero_f32(float* __restrict__ p, int n) {
  int i = blockIdx.x * blockDim.x + threadIdx.x;
  if (i < n) p[i] = 0.0f;
}

__global__ void f32_to_bf16(const float* __restrict__ in,
                            bf16_t* __restrict__ out, int n) {
  int i = blockIdx.x * blockDim.x + threadIdx.x;
  if (i < n) out[i] = (bf16_t)in[i];
}

// ---------------------------------------------------------------------------
// Decode: A_pred = sigmoid(Z @ Z^T), Z is 16384x32 bf16 (row-major, 64 B rows).
// One wave handles a 16 x (16*JT) strip: A fragment loaded once, JT
// independent v_wmma_f32_16x16x32_bf16 (K=32 == one WMMA per tile).
// Sigmoid via v_exp_f32 + v_rcp_f32 (fast rcp: avoids the IEEE div sequence).
// 1 GiB output streamed with non-temporal stores; HBM floor ~46us @ 23.3TB/s.
// All element offsets < 2^28, so 32-bit offset math off a uniform base.
// ---------------------------------------------------------------------------
__global__ void decode_sigmoid_wmma(const bf16_t* __restrict__ Zb,
                                    float* __restrict__ out,
                                    int ntile, int N) {
  const unsigned lane   = threadIdx.x;
  const unsigned waveId = blockIdx.x * WPB + threadIdx.y;
  const unsigned gpr    = (unsigned)ntile / JT;    // tile-groups per row
  const unsigned itile  = waveId / gpr;
  const unsigned jt0    = (waveId - itile * gpr) * JT;

  const unsigned half = lane >> 4;
  const unsigned l15  = lane & 15;

  // A fragment: row itile*16+l15 of Z; two contiguous 16 B chunks per lane.
  const bf16_t* ar = Zb + (itile * 16 + l15) * 32u;
  v8bf alo = *(const v8bf*)(ar + half * 16);
  v8bf ahi = *(const v8bf*)(ar + half * 16 + 8);
  v16bf a = __builtin_shufflevector(alo, ahi, 0, 1, 2, 3, 4, 5, 6, 7,
                                    8, 9, 10, 11, 12, 13, 14, 15);

  const unsigned obase = (itile * 16 + half * 8) * (unsigned)N + l15;
#pragma unroll
  for (unsigned q = 0; q < JT; ++q) {
    const unsigned jtile = jt0 + q;
    // B fragment: B(k,n) = Z[jtile*16+n][k]; one contiguous 32 B chunk/lane.
    const bf16_t* br = Zb + (jtile * 16 + l15) * 32u;
    v16bf b = *(const v16bf*)(br + half * 16);

    v8f acc = {};
    acc = __builtin_amdgcn_wmma_f32_16x16x32_bf16(false, a, false, b,
                                                  (short)0, acc, false, false);

    const unsigned off = obase + jtile * 16;
#pragma unroll
    for (unsigned r = 0; r < 8; ++r) {
      float ex = __expf(-acc[r]);
      float v  = __builtin_amdgcn_rcpf(1.0f + ex);   // v_rcp_f32, not IEEE div
      __builtin_nontemporal_store(v, out + off + r * (unsigned)N);
    }
  }
}

// ---------------------------------------------------------------------------
extern "C" void kernel_launch(void* const* d_in, const int* in_sizes, int n_in,
                              void* d_out, int out_size, void* d_ws, size_t ws_size,
                              hipStream_t stream) {
  const float*     X   = (const float*)d_in[0];
  const long long* ei  = (const long long*)d_in[1];   // int64 (2, E)
  const float*     ew  = (const float*)d_in[2];
  const float*     W1  = (const float*)d_in[3];
  const float*     W2  = (const float*)d_in[4];
  // d_in[5] = t (unused by reference), d_in[6] = adj_distance (unused)

  const int F  = 512, H1 = 64, H2 = 32;
  const int E  = in_sizes[2];                 // 524288
  const int N  = in_sizes[0] / F;             // 16384 nodes
  const long long* srcIdx = ei;
  const long long* dstIdx = ei + E;

  // Workspace layout (8 MB total, reused):
  //   buf0: N*H1 fp32 (4 MB): H = X@W1, then G = relu(Hagg)@W2, then Zb (bf16)
  //   buf1: N*H1 fp32 (4 MB): Hagg, then Z
  float*  buf0 = (float*)d_ws;
  float*  buf1 = buf0 + (size_t)N * H1;
  bf16_t* Zb   = (bf16_t*)buf0;               // overlays dead G after spmm2

  float* Apred = (float*)d_out;
  dim3 wblk(32, WPB, 1);

  // 1) H = X @ W1    (16384x512 @ 512x64) -> buf0
  {
    int tiles = (N / 16) * (H1 / 16);         // 4096 waves
    gemm16_bf16_wmma<false><<<tiles / WPB, wblk, 0, stream>>>(X, W1, buf0, N, H1, F);
  }
  // 2) Hagg = spmm(A, H) -> buf1 (zero first; atomics accumulate)
  {
    int n = N * H1;
    zero_f32<<<(n + 255) / 256, 256, 0, stream>>>(buf1, n);
    long long thr = (long long)E * H1;
    spmm_scatter<6><<<(int)(thr / 256), 256, 0, stream>>>(srcIdx, dstIdx, ew,
                                                          buf0, buf1, E);
  }
  // 3) G = relu(Hagg) @ W2   (16384x64 @ 64x32) -> buf0  (ReLU fused into A load)
  {
    int tiles = (N / 16) * (H2 / 16);         // 2048 waves
    gemm16_bf16_wmma<true><<<tiles / WPB, wblk, 0, stream>>>(buf1, W2, buf0, N, H2, H1);
  }
  // 4) Z = spmm(A, G) -> buf1 (zero first)
  {
    int n = N * H2;
    zero_f32<<<(n + 255) / 256, 256, 0, stream>>>(buf1, n);
    long long thr = (long long)E * H2;
    spmm_scatter<5><<<(int)(thr / 256), 256, 0, stream>>>(srcIdx, dstIdx, ew,
                                                          buf0, buf1, E);
  }
  // 5) Zb = bf16(Z)   (row-major 16384x32, 64 B rows)
  {
    int n = N * H2;
    f32_to_bf16<<<(n + 255) / 256, 256, 0, stream>>>(buf1, Zb, n);
  }
  // 6) A_pred = sigmoid(Z @ Z^T) — 1024x1024 tiles, JT tiles per wave,
  //    one WMMA per tile, NT stores.
  {
    int ntile = N / 16;                       // 1024
    long long waves = ((long long)ntile * ntile) / JT;
    decode_sigmoid_wmma<<<(int)(waves / WPB), wblk, 0, stream>>>(Zb, Apred, ntile, N);
  }
}
